// TextInputAutoregressive_34179349742328
// MI455X (gfx1250) — matrix operand order repaired and checked by reading, not compiled, not run on hardware
//
#include <hip/hip_runtime.h>
#include <hip/hip_bf16.h>

// ---------------------------------------------------------------------------
// y_t = y_{t-1} @ M^T + x_t  autoregressive scan on MI455X (gfx1250, wave32)
//
// Design: latency-bound sequential recurrence -> single persistent workgroup
// (32 waves on one WGP). y kept in LDS pre-packed in WMMA A-fragment layout
// (double-buffered: ONE barrier per step); M kept in L2 as bf16; f32
// accumulate via V_WMMA_F32_16X16X32_BF16.
// ---------------------------------------------------------------------------

typedef __attribute__((ext_vector_type(16))) __bf16 v16bf;
typedef __attribute__((ext_vector_type(4)))  __bf16 v4bf;
typedef __attribute__((ext_vector_type(8)))  float  v8f;

#define D_MODEL 1024
#define BATCH   4
#define SEQ     2048
#define SEQ_OUT (SEQ + 1)
#define AFRAG_ELEMS (32 * 32 * 16)   // [kt][lane][slot] bf16 -> 32 KB per buffer

// ---- kernel 1: M_w fp32 -> bf16, row-major [n][k] (B[k][n] contiguous in k)
__global__ void k_convert_M(const float* __restrict__ Mw, __bf16* __restrict__ Mbf) {
    int i = (blockIdx.x * blockDim.x + threadIdx.x) * 4;
    if (i >= D_MODEL * D_MODEL) return;
    float4 v = *reinterpret_cast<const float4*>(Mw + i);
    v4bf o = { (__bf16)v.x, (__bf16)v.y, (__bf16)v.z, (__bf16)v.w };
    *reinterpret_cast<v4bf*>(Mbf + i) = o;
}

// ---- kernel 2: embedding gather, out[b][t][:] = emb[padded_id(b,t)][:]
// out also serves as x_t storage; scan updates rows t>=1 in place.
__global__ void k_embed(const int* __restrict__ ids, const float* __restrict__ emb,
                        float* __restrict__ out) {
    int tid = blockIdx.x * blockDim.x + threadIdx.x;   // one thread per float4
    int d4  = tid & 255;                               // 256 float4 per row
    int bt  = tid >> 8;
    if (bt >= BATCH * SEQ_OUT) return;
    int t = bt % SEQ_OUT;
    int b = bt / SEQ_OUT;
    int id = (t == 0) ? 0 : ids[b * SEQ + (t - 1)];
    float4 v = *reinterpret_cast<const float4*>(emb + (size_t)id * D_MODEL + d4 * 4);
    *reinterpret_cast<float4*>(out + ((size_t)bt) * D_MODEL + d4 * 4) = v;
}

// ---- A-fragment scatter helper: y element (row b, global col k) -> LDS slot
// A 16x32 packing used here (not AMD's table order): lanes 0-15 hold K {0..15}; lanes 16-31 hold
// K {16..31} of the same rows (contiguous, matching B fragments).
__device__ __forceinline__ void afrag_store(__bf16* afrag, int b, int k, float v) {
    int kt   = k >> 5;
    int kw   = k & 31;
    int hh   = kw >> 4;                       // which half-wave holds it
    int slot = kw & 15;                       // 0..15
    afrag[((kt * 32) + (b + (hh << 4))) * 16 + slot] = (__bf16)v;
}

// ---- kernel 3: persistent scan. One workgroup = 32 waves; wave w owns output
// columns [32w, 32w+32) as two 16x16 C tiles; K-loop of 32 WMMAs per tile.
// A fragments double-buffered in LDS: step t reads buf[(t-1)&1], writes
// buf[t&1]; a single barrier per step publishes the writes and protects the
// buffer that gets overwritten next step.
__global__ __launch_bounds__(1024, 1)
void k_scan(const __bf16* __restrict__ Mbf, float* __restrict__ out) {
    __shared__ alignas(32) __bf16 afrag[2][AFRAG_ELEMS];   // 64 KB of 320 KB

    const int tid  = threadIdx.x;
    const int lane = tid & 31;
    const int wave = tid >> 5;
    const int r16  = lane & 15;
    const int half = lane >> 4;

    // Build A fragments for y0 = x0 (= out[:,0,:]) into buffer 0.
    {
        const int k = tid;
        #pragma unroll
        for (int b = 0; b < BATCH; ++b) {
            float v = out[((size_t)b * SEQ_OUT) * D_MODEL + k];
            afrag_store(afrag[0], b, k, v);
        }
    }
    __syncthreads();

    const int c0col = wave * 32 + r16;        // tile 0 column (also N index)
    const int c1col = wave * 32 + 16 + r16;   // tile 1 column
    // B fragment: lane (N=r16, K-half=half) reads 16 contiguous bf16 along k.
    const __bf16* brow0 = Mbf + (size_t)c0col * D_MODEL + half * 16;
    const __bf16* brow1 = Mbf + (size_t)c1col * D_MODEL + half * 16;

    for (int t = 1; t <= SEQ; ++t) {
        const __bf16* ard = afrag[(t - 1) & 1];
        __bf16*       awr = afrag[t & 1];

        // Init accumulators with x_t (C layout: VGPR r, lane l -> M=r+8*half, N=l&15).
        v8f c0 = {}, c1 = {};
        if (half == 0) {
            #pragma unroll
            for (int b = 0; b < BATCH; ++b) {
                c0[b] = out[((size_t)b * SEQ_OUT + t) * D_MODEL + c0col];
                c1[b] = out[((size_t)b * SEQ_OUT + t) * D_MODEL + c1col];
            }
        }

        #pragma unroll 8
        for (int kt = 0; kt < 32; ++kt) {
            v16bf a  = *reinterpret_cast<const v16bf*>(&ard[(kt * 32 + lane) * 16]);
            v16bf b0 = *reinterpret_cast<const v16bf*>(brow0 + kt * 32);
            v16bf b1 = *reinterpret_cast<const v16bf*>(brow1 + kt * 32);
            c0 = __builtin_amdgcn_wmma_f32_16x16x32_bf16(false, a, false, b0,
                                                         (short)0, c0, false, false);
            c1 = __builtin_amdgcn_wmma_f32_16x16x32_bf16(false, a, false, b1,
                                                         (short)0, c1, false, false);
        }

        // Write y_t to global out and repack bf16 into the other LDS buffer.
        if (half == 0) {
            #pragma unroll
            for (int b = 0; b < BATCH; ++b) {
                float v0 = c0[b];
                out[((size_t)b * SEQ_OUT + t) * D_MODEL + c0col] = v0;
                afrag_store(awr, b, c0col, v0);
                float v1 = c1[b];
                out[((size_t)b * SEQ_OUT + t) * D_MODEL + c1col] = v1;
                afrag_store(awr, b, c1col, v1);
            }
        }
        __syncthreads();   // publish awr; also fences reuse of ard next step
    }
}

extern "C" void kernel_launch(void* const* d_in, const int* in_sizes, int n_in,
                              void* d_out, int out_size, void* d_ws, size_t ws_size,
                              hipStream_t stream) {
    const int*   ids = (const int*)d_in[0];    // input_ids (BATCH, SEQ)
    const float* emb = (const float*)d_in[1];  // (N_VOCAB, D_MODEL)
    const float* Mw  = (const float*)d_in[2];  // (D_MODEL, D_MODEL)
    float* out  = (float*)d_out;               // (BATCH, SEQ_OUT, D_MODEL)
    __bf16* Mbf = (__bf16*)d_ws;               // 2 MB scratch, fits L2

    {   // fp32 -> bf16 weight convert
        int n = (D_MODEL * D_MODEL) / 4;
        k_convert_M<<<(n + 255) / 256, 256, 0, stream>>>(Mw, Mbf);
    }
    {   // embedding gather (also writes y0 row)
        int n = (BATCH * SEQ_OUT * D_MODEL) / 4;
        k_embed<<<(n + 255) / 256, 256, 0, stream>>>(ids, emb, out);
    }
    // persistent single-WGP recurrence
    k_scan<<<1, 1024, 0, stream>>>(Mbf, out);
}
